// CausalSelfAttention_77695958385275
// MI455X (gfx1250) — matrix operand order repaired and checked by reading, not compiled, not run on hardware
//
#include <hip/hip_runtime.h>
#include <hip/hip_bf16.h>

typedef __attribute__((ext_vector_type(16))) _Float16 v16h;
typedef __attribute__((ext_vector_type(8)))  float    v8f;

#define B_        2
#define T_        2048
#define DIM_      2048
#define H_        16
#define KV_       8
#define HD_       128
#define NUM_FULL_ 8
#define STRIDE_   45
#define NKS_      46                      // number of strided key positions
#define QKV_N     ((H_ + 2*KV_)*HD_)      // 4096
#define SCALE_    0.08838834764831845f    // 1/sqrt(128)

__device__ __forceinline__ v8f wmma16x16x32(v16h a, v16h b, v8f c) {
  return __builtin_amdgcn_wmma_f32_16x16x32_f16(false, a, false, b, (short)0, c,
                                                false, false);
}

// A fragment (16x32 f16): lane m=lane&15, half=lane>>4, K(i)=16*half+i (contiguous)
__device__ __forceinline__ v16h load_a_f32(const float* src, int ld, int lane) {
  int m = lane & 15, hf = lane >> 4;
  const float* p = src + (size_t)m*ld + hf*16;
  v16h a;
#pragma unroll
  for (int g = 0; g < 2; ++g)
#pragma unroll
    for (int j = 0; j < 8; ++j) a[g*8+j] = (_Float16)p[g*8 + j];
  return a;
}
__device__ __forceinline__ v16h load_a_f16(const _Float16* src, int ld, int lane) {
  int m = lane & 15, hf = lane >> 4;
  const _Float16* p = src + (size_t)m*ld + hf*16;
  v16h a;
#pragma unroll
  for (int g = 0; g < 2; ++g)
#pragma unroll
    for (int j = 0; j < 8; ++j) a[g*8+j] = p[g*8 + j];
  return a;
}
// B fragment (32x16): lane holds column n=lane&15; K(i)=16*half+i (contiguous).
// Source is row-major [N, K] (i.e. C = A * W^T): row n is K-contiguous.
__device__ __forceinline__ v16h load_b_f32(const float* w, int ld, int lane) {
  int n = lane & 15, hf = lane >> 4;
  const float* p = w + (size_t)n*ld + hf*16;
  v16h b;
#pragma unroll
  for (int i = 0; i < 16; ++i) b[i] = (_Float16)p[i];
  return b;
}

// per-lane async copy of 16 bytes global -> LDS (ASYNCcnt tracked)
__device__ __forceinline__ void async_cp16(unsigned lds_off, const void* gptr) {
  unsigned long long ga = (unsigned long long)(uintptr_t)gptr;
  asm volatile("global_load_async_to_lds_b128 %0, %1, off"
               :: "v"(lds_off), "v"(ga) : "memory");
}
__device__ __forceinline__ void async_wait0() {
  asm volatile("s_wait_asynccnt 0x0" ::: "memory");
}

// ------------- C[M,N] = A[M,K] * W[N,K]^T, block-cooperative LDS staging -------------
// Block = 256 threads = 8 waves (4 M-waves x 2 N-waves); block tile 128x128,
// per-wave tile 32x64, K-step 32. Tiles staged via global_load_async_to_lds_b128.
// Padded LDS rows (36 f32 / 40 f16) keep 16B alignment and avoid bank conflicts.
template <bool AHALF>
__global__ __launch_bounds__(256)
void gemm_awt(const void* __restrict__ Ap, const float* __restrict__ W,
              float* __restrict__ C, int M, int N, int K) {
  constexpr int AROW   = AHALF ? 40 : 36;                 // elements per staged A row
  constexpr int BROW   = 36;                              // f32 elements per staged B row
  constexpr int ABYTES = 128 * AROW * (AHALF ? 2 : 4);    // 10240 or 18432
  constexpr int BBYTES = 128 * BROW * 4;                  // 18432
  __shared__ __align__(16) char smem[ABYTES + BBYTES];
  char* aS = smem;
  char* bS = smem + ABYTES;
  unsigned aOff = (unsigned)(uintptr_t)aS;   // LDS byte offset = low 32 bits
  unsigned bOff = (unsigned)(uintptr_t)bS;

  int tid  = threadIdx.x;
  int lane = tid & 31;
  int warp = tid >> 5;
  int wy = warp >> 1, wx = warp & 1;         // 4 x 2 wave grid
  int tilesN = N >> 7;
  int bm = blockIdx.x / tilesN, bn = blockIdx.x - bm * tilesN;
  int rowB = bm << 7, colB = bn << 7;

  v8f acc[2][4];
#pragma unroll
  for (int i = 0; i < 2; ++i)
#pragma unroll
    for (int j = 0; j < 4; ++j) acc[i][j] = (v8f)(0.0f);

  for (int k0 = 0; k0 < K; k0 += 32) {
    // ---- stage A tile (128 rows x 32 k) ----
    if (AHALF) {
      const _Float16* A = (const _Float16*)Ap;
      int r = tid >> 2, c = (tid & 3) * 8;                 // 8 f16 = 16B per lane
#pragma unroll
      for (int i = 0; i < 2; ++i) {
        int row = r + i * 64;
        async_cp16(aOff + (unsigned)(row * AROW + c) * 2,
                   A + (size_t)(rowB + row) * K + k0 + c);
      }
    } else {
      const float* A = (const float*)Ap;
      int r = tid >> 3, c = (tid & 7) * 4;                 // 4 f32 = 16B per lane
#pragma unroll
      for (int i = 0; i < 4; ++i) {
        int row = r + i * 32;
        async_cp16(aOff + (unsigned)(row * AROW + c) * 4,
                   A + (size_t)(rowB + row) * K + k0 + c);
      }
    }
    // ---- stage B tile (128 W-rows x 32 k, f32) ----
    {
      int r = tid >> 3, c = (tid & 7) * 4;
#pragma unroll
      for (int i = 0; i < 4; ++i) {
        int row = r + i * 32;
        async_cp16(bOff + (unsigned)(row * BROW + c) * 4,
                   W + (size_t)(colB + row) * K + k0 + c);
      }
    }
    async_wait0();
    __syncthreads();

    // ---- compute from LDS ----
    v16h a0, a1;
    if (AHALF) {
      const _Float16* aT = (const _Float16*)aS;
      a0 = load_a_f16(aT + (size_t)(wy*32)      * AROW, AROW, lane);
      a1 = load_a_f16(aT + (size_t)(wy*32 + 16) * AROW, AROW, lane);
    } else {
      const float* aT = (const float*)aS;
      a0 = load_a_f32(aT + (size_t)(wy*32)      * AROW, AROW, lane);
      a1 = load_a_f32(aT + (size_t)(wy*32 + 16) * AROW, AROW, lane);
    }
    const float* bT = (const float*)bS;
#pragma unroll
    for (int j = 0; j < 4; ++j) {
      v16h b = load_b_f32(bT + (size_t)(wx*64 + j*16) * BROW, BROW, lane);
      acc[0][j] = wmma16x16x32(a0, b, acc[0][j]);
      acc[1][j] = wmma16x16x32(a1, b, acc[1][j]);
    }
    __syncthreads();                                       // before next overwrite
  }

  // C/D layout: element r -> row r+8*half, col lane&15
  int n = lane & 15, hf = lane >> 4;
  int row0 = rowB + wy*32, col0 = colB + wx*64;
#pragma unroll
  for (int i = 0; i < 2; ++i)
#pragma unroll
    for (int j = 0; j < 4; ++j)
#pragma unroll
      for (int r = 0; r < 8; ++r)
        C[(size_t)(row0 + i*16 + r + 8*hf)*N + col0 + j*16 + n] = acc[i][j][r];
}

// ---------------- RoPE + gain + pack to f16 ----------------
__global__ __launch_bounds__(256)
void rope_pack(const float* __restrict__ qkv, const float* __restrict__ q_gain,
               const float* __restrict__ attn_temp,
               _Float16* __restrict__ qb, _Float16* __restrict__ kb,
               _Float16* __restrict__ vb) {
  size_t idx = (size_t)blockIdx.x * blockDim.x + threadIdx.x;
  if (idx >= (size_t)B_ * T_ * QKV_N) return;
  int    c  = (int)(idx % QKV_N);
  size_t bt = idx / QKV_N;
  int    t  = (int)(bt % T_);
  int    b  = (int)(bt / T_);
  const float* row = qkv + bt * QKV_N;

  if (c < H_*HD_) {                              // Q heads: rope + gain + SCALE
    int hh = c / HD_, d = c % HD_;
    float val;
    if (d < 64) {
      float x1 = row[c], x2 = row[c + 64];
      float ang = (float)t * ((float)d * (1.0f/64.0f));
      val = x1*cosf(ang) - x2*sinf(ang);
    } else {
      float x1 = row[c - 64], x2 = row[c];
      float ang = (float)t * ((float)(d-64) * (1.0f/64.0f));
      val = x1*sinf(ang) + x2*cosf(ang);
    }
    val *= q_gain[hh] * attn_temp[hh] * SCALE_;
    qb[((size_t)(b*H_ + hh)*T_ + t)*HD_ + d] = (_Float16)val;
  } else if (c < (H_+KV_)*HD_) {                 // K heads: rope
    int cc = c - H_*HD_;
    int hh = cc / HD_, d = cc % HD_;
    float val;
    if (d < 64) {
      float x1 = row[c], x2 = row[c + 64];
      float ang = (float)t * ((float)d * (1.0f/64.0f));
      val = x1*cosf(ang) - x2*sinf(ang);
    } else {
      float x1 = row[c - 64], x2 = row[c];
      float ang = (float)t * ((float)(d-64) * (1.0f/64.0f));
      val = x1*sinf(ang) + x2*cosf(ang);
    }
    kb[((size_t)(b*KV_ + hh)*T_ + t)*HD_ + d] = (_Float16)val;
  } else {                                       // V heads: copy
    int cc = c - (H_+KV_)*HD_;
    int hh = cc / HD_, d = cc % HD_;
    vb[((size_t)(b*KV_ + hh)*T_ + t)*HD_ + d] = (_Float16)row[c];
  }
}

// ---------------- Flash attention: one wave = (b, h, 16-query tile) ----------------
__global__ __launch_bounds__(256)
void attn_kernel(const _Float16* __restrict__ qb, const _Float16* __restrict__ kb,
                 const _Float16* __restrict__ vb, _Float16* __restrict__ yb) {
  __shared__ _Float16 ldsp[8][16*32];            // per-wave P tile (row-major 16x32)
  int lane  = threadIdx.x & 31;
  int wslot = threadIdx.x >> 5;
  int wid   = (int)((blockIdx.x * blockDim.x + threadIdx.x) >> 5);
  const int QT = T_/16;
  int qt = wid % QT;
  int bh = wid / QT;
  int h  = bh % H_;
  int b  = bh / H_;
  if (b >= B_) return;                           // wave-uniform, grid exact
  int  kvh    = h >> 1;                          // GQA 2:1
  bool sparse = (h >= NUM_FULL_);
  _Float16* pb = ldsp[wslot];
  int n = lane & 15, hf = lane >> 4;

  // Q fragments (SCALE and gains already folded into q)
  const _Float16* qrow = qb + ((size_t)(b*H_ + h)*T_ + qt*16)*HD_;
  v16h aq[4];
#pragma unroll
  for (int kk = 0; kk < 4; ++kk) aq[kk] = load_a_f16(qrow + kk*32, HD_, lane);

  const _Float16* kbase = kb + (size_t)(b*KV_ + kvh)*T_*HD_;
  const _Float16* vbase = vb + (size_t)(b*KV_ + kvh)*T_*HD_;

  v8f o[8];
#pragma unroll
  for (int f = 0; f < 8; ++f) o[f] = (v8f)(0.0f);
  float mrow[8], lrow[8];
#pragma unroll
  for (int r = 0; r < 8; ++r) { mrow[r] = -1e30f; lrow[r] = 0.0f; }

  int qmax  = qt*16 + 15;
  int nkeys = sparse ? NKS_ : (qmax + 1);
  for (int j0 = 0; j0 < nkeys; j0 += 32) {
    // ---- S = Q * K^T for 32 keys (two 16-key fragments) ----
    v8f s0 = (v8f)(0.0f), s1 = (v8f)(0.0f);
    int jc0 = j0 + n, jc1 = j0 + 16 + n;
    int tk0 = sparse ? ((jc0 < NKS_ ? jc0 : NKS_-1) * STRIDE_) : jc0;
    int tk1 = sparse ? ((jc1 < NKS_ ? jc1 : NKS_-1) * STRIDE_) : jc1;
#pragma unroll
    for (int kk = 0; kk < 4; ++kk) {             // reduce over HD in steps of 32
      const _Float16* p0 = kbase + (size_t)tk0*HD_ + kk*32 + hf*16;
      const _Float16* p1 = kbase + (size_t)tk1*HD_ + kk*32 + hf*16;
      v16h b0, b1;
#pragma unroll
      for (int i = 0; i < 16; ++i) { b0[i] = p0[i]; b1[i] = p1[i]; }
      s0 = wmma16x16x32(aq[kk], b0, s0);
      s1 = wmma16x16x32(aq[kk], b1, s1);
    }
    // ---- mask + online softmax (row = element r + 8*half, cols across lanes) ----
#pragma unroll
    for (int r = 0; r < 8; ++r) {
      int  qpos = qt*16 + r + 8*hf;
      int  kp0  = sparse ? jc0*STRIDE_ : jc0;
      int  kp1  = sparse ? jc1*STRIDE_ : jc1;
      bool ok0  = (qpos >= kp0) && (!sparse || jc0 < NKS_);
      bool ok1  = (qpos >= kp1) && (!sparse || jc1 < NKS_);
      float sv0 = ok0 ? s0[r] : -1e30f;
      float sv1 = ok1 ? s1[r] : -1e30f;
      float pm  = fmaxf(sv0, sv1);
#pragma unroll
      for (int off = 1; off < 16; off <<= 1) pm = fmaxf(pm, __shfl_xor(pm, off, 32));
      float mnew = fmaxf(mrow[r], pm);
      float sc   = __expf(mrow[r] - mnew);
      float p0   = __expf(sv0 - mnew);
      float p1   = __expf(sv1 - mnew);
      float ps   = p0 + p1;
#pragma unroll
      for (int off = 1; off < 16; off <<= 1) ps += __shfl_xor(ps, off, 32);
      lrow[r] = lrow[r]*sc + ps;
      mrow[r] = mnew;
      pb[(r + 8*hf)*32 + n]      = (_Float16)p0;   // P tile row-major in LDS
      pb[(r + 8*hf)*32 + 16 + n] = (_Float16)p1;
#pragma unroll
      for (int f = 0; f < 8; ++f) o[f][r] *= sc;   // rescale running O
    }
    // ---- O += P * V : P re-read from LDS in A-fragment layout ----
    v16h ap = load_a_f16(pb, 32, lane);
#pragma unroll
    for (int f = 0; f < 8; ++f) {                // HD in 16-wide column tiles
      v16h bv;
#pragma unroll
      for (int i = 0; i < 16; ++i) {             // K(i)=16*half+i over keys
        int key = j0 + hf*16 + i;
        int tk  = sparse ? ((key < NKS_ ? key : NKS_-1) * STRIDE_) : key;
        bv[i] = vbase[(size_t)tk*HD_ + f*16 + n];
      }
      o[f] = wmma16x16x32(ap, bv, o[f]);
    }
  }
  // ---- epilogue: divide by row sum, write y as [B,T,H,HD] f16 ----
#pragma unroll
  for (int r = 0; r < 8; ++r) {
    float inv = 1.0f / lrow[r];
    int t = qt*16 + r + 8*hf;
    _Float16* yrow = yb + ((size_t)(b*T_ + t)*H_ + h)*HD_;
#pragma unroll
    for (int f = 0; f < 8; ++f) yrow[f*16 + n] = (_Float16)((float)o[f][r] * inv);
  }
}

extern "C" void kernel_launch(void* const* d_in, const int* in_sizes, int n_in,
                              void* d_out, int out_size, void* d_ws, size_t ws_size,
                              hipStream_t stream) {
  const float* x         = (const float*)d_in[0];
  const float* w_attn    = (const float*)d_in[1];
  const float* w_proj    = (const float*)d_in[2];
  const float* q_gain    = (const float*)d_in[3];
  const float* attn_temp = (const float*)d_in[4];
  float* out = (float*)d_out;

  char* ws = (char*)d_ws;
  float*    qkv  = (float*)ws;                       // 64 MiB [B*T, 4096] f32
  _Float16* ybuf = (_Float16*)ws;                    // 16 MiB, reuses qkv region
  _Float16* qbuf = (_Float16*)(ws + (size_t)(64u << 20));   // 16 MiB [B,H,T,HD]
  _Float16* kbuf = (_Float16*)(ws + (size_t)(80u << 20));   //  8 MiB [B,KV,T,HD]
  _Float16* vbuf = (_Float16*)(ws + (size_t)(88u << 20));   //  8 MiB [B,KV,T,HD]

  // 1) qkv = x @ w_attn^T : M=4096 N=4096 K=2048, 128x128 block tiles
  {
    int M = B_*T_, N = QKV_N, K = DIM_;
    gemm_awt<false><<<(M/128)*(N/128), 256, 0, stream>>>(x, w_attn, qkv, M, N, K);
  }
  // 2) RoPE + gain + pack to f16
  {
    size_t tot = (size_t)B_ * T_ * QKV_N;
    rope_pack<<<(int)((tot + 255)/256), 256, 0, stream>>>(qkv, q_gain, attn_temp,
                                                          qbuf, kbuf, vbuf);
  }
  // 3) attention (full + strided-sparse heads)
  {
    int waves = B_ * H_ * (T_/16);                   // 4096 waves
    attn_kernel<<<waves/8, 256, 0, stream>>>(qbuf, kbuf, vbuf, ybuf);
  }
  // 4) out = y @ w_proj^T : M=4096 N=2048 K=2048
  {
    int M = B_*T_, N = DIM_, K = DIM_;
    gemm_awt<true><<<(M/128)*(N/128), 256, 0, stream>>>(ybuf, w_proj, out, M, N, K);
  }
}
